// LSTMOnly_72335839199444
// MI455X (gfx1250) — matrix operand order repaired and checked by reading, not compiled, not run on hardware
//
#include <hip/hip_runtime.h>

typedef __attribute__((ext_vector_type(8)))  _Float16 v8h;
typedef __attribute__((ext_vector_type(16))) _Float16 v16h;
typedef __attribute__((ext_vector_type(8)))  float    v8f;

constexpr int BATCH = 256;
constexpr int SEQ   = 1024;
constexpr int H1D   = 128;
constexpr int H2D   = 64;
constexpr int NCLS  = 4;
constexpr int XCH   = 32;   // x prefetch chunk (timesteps)

// CDNA5 TRANS-unit tanh. Prefer the builtin (compiler handles the TRANS
// result hazard); fall back to inline asm with a trailing v_nop per ISA 7.4.
__device__ __forceinline__ float tanh_hw(float x) {
#if __has_builtin(__builtin_amdgcn_tanhf)
  return __builtin_amdgcn_tanhf(x);
#else
  float r;
  asm("v_tanh_f32 %0, %1\n\tv_nop" : "=v"(r) : "v"(x));
  return r;
#endif
}
__device__ __forceinline__ float sigm(float x) {
  return __builtin_fmaf(0.5f, tanh_hw(0.5f * x), 0.5f);
}

// Assemble a v16h fragment from two 8-half (16B) chunks.
__device__ __forceinline__ v16h ld_frag(const _Float16* p0, const _Float16* p1) {
  v8h lo = *(const v8h*)p0;
  v8h hi = *(const v8h*)p1;
  v16h r;
#pragma unroll
  for (int i = 0; i < 8; ++i) { r[i] = lo[i]; r[i + 8] = hi[i]; }
  return r;
}

// A-matrix fragment (16x32 f16) from row-major f16, stride in halves.
// ISA 7.12.2: lane m=lane&15, hi=lane>>4; elems 0..7 -> K=kc*32+hi*8+j,
// elems 8..15 -> K=kc*32+16+hi*8+j.
__device__ __forceinline__ v16h ld_a(const _Float16* base, int strideH, int kc, int lane) {
  int m  = lane & 15;
  int hi = lane >> 4;
  const _Float16* p0 = base + (size_t)m * strideH + kc * 32 + hi * 8;
  return ld_frag(p0, p0 + 16);
}

__device__ __forceinline__ v8f wmma16(v16h a, v16h b, v8f c) {
  return __builtin_amdgcn_wmma_f32_16x16x32_f16(false, a, false, b, (short)0, c, false, false);
}

// Row-major (N x K) f32 -> lane-swizzled f16 B-fragments.
// interleave=0: tile nt covers rows nt*16..nt*16+15 (linear).
// interleave=1: tile nt -> gate g=nt&3, jtile jt=nt>>2; rows n = g*(N/4) + jt*16 + lane15
//   so one wave's 4 consecutive tiles are the (i,f,g,o) rows of the SAME hidden columns.
__global__ void k_swz(const float* __restrict__ w, _Float16* __restrict__ dst,
                      int N, int K, int interleave) {
  int idx = blockIdx.x * blockDim.x + threadIdx.x;
  if (idx >= N * K) return;
  int j    = idx & 15;
  int lane = (idx >> 4) & 31;
  int rest = idx >> 9;
  int KC   = K >> 5;
  int kc   = rest % KC;
  int nt   = rest / KC;
  int n;
  if (interleave) {
    int g = nt & 3, jt = nt >> 2;
    n = g * (N >> 2) + jt * 16 + (lane & 15);
  } else {
    n = nt * 16 + (lane & 15);
  }
  int k = kc * 32 + (lane >> 4) * 8 + (j & 7) + (j >> 3) * 16;
  dst[idx] = (_Float16)w[n * K + k];
}

// ---------------- Layer 1: persistent bidirectional recurrence ----------------
// grid (16, 2): batch chunk of 16 x direction. 256 threads = 8 waves.
// Wave w owns hidden cols [w*16, w*16+16) with all 4 gate tiles in registers;
// nonlinearities + c/h update are fully in-register. One barrier per step
// (double-buffered h in LDS).
__global__ __launch_bounds__(256) void k_lstm1(
    const float* __restrict__ x,
    const _Float16* __restrict__ whh_f, const _Float16* __restrict__ whh_b,
    const float* __restrict__ wih_f, const float* __restrict__ wih_b,
    const float* __restrict__ bih_f, const float* __restrict__ bih_b,
    const float* __restrict__ bhh_f, const float* __restrict__ bhh_b,
    _Float16* __restrict__ h1out)   // [t][b][256] f16; fwd cols 0..127, bwd 128..255
{
  __shared__ _Float16 hbuf[2][16][H1D];   // 8 KB double-buffered h state
  __shared__ float    xchunk[16][XCH];    // 2 KB

  const int tid  = threadIdx.x;
  const int wave = tid >> 5;
  const int lane = tid & 31;
  const int l15  = lane & 15;
  const int hi   = lane >> 4;
  const int bm   = blockIdx.x * 16;
  const int dir  = blockIdx.y;
  const int jcol = wave * 16 + l15;

  const _Float16* whh = dir ? whh_b : whh_f;
  const float* wih = dir ? wih_b : wih_f;
  const float* bih = dir ? bih_b : bih_f;
  const float* bhh = dir ? bhh_b : bhh_f;

  // Persistent recurrent weights: 4 gate tiles x 4 K-chunks, VGPR-resident.
  v16h bw[4][4];
#pragma unroll
  for (int g = 0; g < 4; ++g)
#pragma unroll
    for (int kc = 0; kc < 4; ++kc) {
      const _Float16* p = whh + ((((wave * 4 + g) * 4 + kc) * 32 + lane) << 4);
      bw[g][kc] = ld_frag(p, p + 8);
    }

  float wxv[4], bsv[4];
#pragma unroll
  for (int g = 0; g < 4; ++g) {
    int n = g * H1D + jcol;      // gate-interleaved row index
    wxv[g] = wih[n];             // w_ih1 is (512,1): rank-1 input projection
    bsv[g] = bih[n] + bhh[n];
  }

  for (int idx = tid; idx < 16 * H1D; idx += 256) hbuf[0][idx >> 7][idx & 127] = (_Float16)0.0f;
  float cst[8];
#pragma unroll
  for (int r = 0; r < 8; ++r) cst[r] = 0.0f;

  for (int s = 0; s < SEQ; ++s) {
    __syncthreads();
    if ((s & (XCH - 1)) == 0) {   // coalesced x prefetch, every 32 steps
      for (int idx = tid; idx < 16 * XCH; idx += 256) {
        int row = idx >> 5, ii = idx & (XCH - 1);
        int ti = dir ? (SEQ - 1 - (s + ii)) : (s + ii);
        xchunk[row][ii] = x[(bm + row) * SEQ + ti];
      }
      __syncthreads();
    }
    const int cur = s & 1;
    const int t = dir ? (SEQ - 1 - s) : s;

    v16h a[4];
#pragma unroll
    for (int kc = 0; kc < 4; ++kc) a[kc] = ld_a(&hbuf[cur][0][0], H1D, kc, lane);

    float xv[8];
#pragma unroll
    for (int r = 0; r < 8; ++r) xv[r] = xchunk[hi * 8 + r][s & (XCH - 1)];

    v8f acc[4];
#pragma unroll
    for (int g = 0; g < 4; ++g) {
#pragma unroll
      for (int r = 0; r < 8; ++r) acc[g][r] = __builtin_fmaf(xv[r], wxv[g], bsv[g]);
#pragma unroll
      for (int kc = 0; kc < 4; ++kc) acc[g] = wmma16(a[kc], bw[g][kc], acc[g]);
    }

    // In-register gate math: lane holds i,f,g,o for (m = hi*8+r, j = jcol).
#pragma unroll
    for (int r = 0; r < 8; ++r) {
      int m = hi * 8 + r;
      float ig = sigm(acc[0][r]);
      float fg = sigm(acc[1][r]);
      float gg = tanh_hw(acc[2][r]);
      float og = sigm(acc[3][r]);
      cst[r] = fg * cst[r] + ig * gg;
      float h = og * tanh_hw(cst[r]);
      hbuf[cur ^ 1][m][jcol] = (_Float16)h;
      h1out[((size_t)t * BATCH + bm + m) * 256 + dir * H1D + jcol] = (_Float16)h;
    }
  }
}

// ---------------- Layer 2 input projection: big parallel WMMA GEMM ----------------
// pre2[(t*B+b)][n] = h1 row (256 f16) @ w_ih2_f^T + b_ih2_f + b_hh2_f   (linear swizzle)
// M = 262144, N = 256, K = 256. grid (2048, 8), 8 waves, each wave 16x32.
__global__ __launch_bounds__(256) void k_pre2(
    const _Float16* __restrict__ h1,
    const _Float16* __restrict__ wswz,
    const float* __restrict__ bih, const float* __restrict__ bhh,
    float* __restrict__ pre2)
{
  const int tid = threadIdx.x, wave = tid >> 5, lane = tid & 31;
  const int hi = lane >> 4;
  const size_t mbase = (size_t)blockIdx.x * 128 + wave * 16;
  const int ntb = blockIdx.y * 2;
  const size_t mrow = mbase + (lane & 15);

  v8f acc0, acc1;
#pragma unroll
  for (int r = 0; r < 8; ++r) { acc0[r] = 0.0f; acc1[r] = 0.0f; }

#pragma unroll
  for (int kc = 0; kc < 8; ++kc) {
    const _Float16* pa = h1 + mrow * 256 + kc * 32 + hi * 8;
    v16h a = ld_frag(pa, pa + 16);
    const _Float16* pb0 = wswz + ((((size_t)ntb * 8 + kc) * 32 + lane) << 4);
    const _Float16* pb1 = wswz + ((((size_t)(ntb + 1) * 8 + kc) * 32 + lane) << 4);
    v16h b0 = ld_frag(pb0, pb0 + 8);
    v16h b1 = ld_frag(pb1, pb1 + 8);
    acc0 = wmma16(a, b0, acc0);
    acc1 = wmma16(a, b1, acc1);
  }
  int n0 = ntb * 16 + (lane & 15), n1 = n0 + 16;
  float bias0 = bih[n0] + bhh[n0], bias1 = bih[n1] + bhh[n1];
#pragma unroll
  for (int r = 0; r < 8; ++r) {
    size_t row = mbase + hi * 8 + r;
    pre2[row * 256 + n0] = acc0[r] + bias0;
    pre2[row * 256 + n1] = acc1[r] + bias1;
  }
}

// ---------------- Layer 2 forward recurrence (persistent, in-register gates) ----------------
// 128 threads = 4 waves; wave w owns hidden cols [w*16, w*16+16), 4 gate tiles.
__global__ __launch_bounds__(128) void k_lstm2_fwd(
    const float* __restrict__ pre2,
    const _Float16* __restrict__ whh,   // gate-interleaved swizzle
    float* __restrict__ h2last)         // [b][128]; fwd writes cols 0..63
{
  __shared__ _Float16 hbuf[2][16][H2D];   // 4 KB

  const int tid = threadIdx.x, wave = tid >> 5, lane = tid & 31;
  const int l15 = lane & 15, hi = lane >> 4;
  const int bm = blockIdx.x * 16;
  const int jcol = wave * 16 + l15;

  v16h bw[4][2];
#pragma unroll
  for (int g = 0; g < 4; ++g)
#pragma unroll
    for (int kc = 0; kc < 2; ++kc) {
      const _Float16* p = whh + ((((wave * 4 + g) * 2 + kc) * 32 + lane) << 4);
      bw[g][kc] = ld_frag(p, p + 8);
    }

  for (int idx = tid; idx < 16 * H2D; idx += 128) hbuf[0][idx >> 6][idx & 63] = (_Float16)0.0f;
  float cst[8];
#pragma unroll
  for (int r = 0; r < 8; ++r) cst[r] = 0.0f;

  // software-pipelined pre2 prefetch (one step ahead)
  float pv[4][8];
#pragma unroll
  for (int g = 0; g < 4; ++g)
#pragma unroll
    for (int r = 0; r < 8; ++r)
      pv[g][r] = pre2[(size_t)(bm + hi * 8 + r) * 256 + g * H2D + jcol];

  for (int t = 0; t < SEQ; ++t) {
    __syncthreads();
    const int cur = t & 1;

    v16h a[2];
#pragma unroll
    for (int kc = 0; kc < 2; ++kc) a[kc] = ld_a(&hbuf[cur][0][0], H2D, kc, lane);

    v8f acc[4];
#pragma unroll
    for (int g = 0; g < 4; ++g)
#pragma unroll
      for (int r = 0; r < 8; ++r) acc[g][r] = pv[g][r];

    if (t + 1 < SEQ) {   // issue next step's pre2 loads; overlap with WMMA + gate math
#pragma unroll
      for (int g = 0; g < 4; ++g)
#pragma unroll
        for (int r = 0; r < 8; ++r)
          pv[g][r] = pre2[((size_t)(t + 1) * BATCH + bm + hi * 8 + r) * 256 + g * H2D + jcol];
    }

#pragma unroll
    for (int g = 0; g < 4; ++g)
#pragma unroll
      for (int kc = 0; kc < 2; ++kc) acc[g] = wmma16(a[kc], bw[g][kc], acc[g]);

#pragma unroll
    for (int r = 0; r < 8; ++r) {
      int m = hi * 8 + r;
      float ig = sigm(acc[0][r]);
      float fg = sigm(acc[1][r]);
      float gg = tanh_hw(acc[2][r]);
      float og = sigm(acc[3][r]);
      cst[r] = fg * cst[r] + ig * gg;
      float h = og * tanh_hw(cst[r]);
      hbuf[cur ^ 1][m][jcol] = (_Float16)h;
      if (t == SEQ - 1) h2last[(bm + m) * 128 + jcol] = h;
    }
  }
}

// ---------------- Layer 2 backward, t = T-1 only (zero initial state) ----------------
// h2_b[:,T-1] = o*tanh(i*g); gates = h1[:,T-1,:] @ w_ih2_b^T + biases (no recurrent matmul).
__global__ __launch_bounds__(128) void k_lstm2_bwd_last(
    const _Float16* __restrict__ h1,
    const _Float16* __restrict__ wswz,  // gate-interleaved swizzle of w_ih2_b
    const float* __restrict__ bih, const float* __restrict__ bhh,
    float* __restrict__ h2last)         // writes cols 64..127
{
  const int tid = threadIdx.x, wave = tid >> 5, lane = tid & 31;
  const int l15 = lane & 15, hi = lane >> 4;
  const int bm = blockIdx.x * 16;
  const int jcol = wave * 16 + l15;
  const size_t mrow = (size_t)(SEQ - 1) * BATCH + bm + l15;

  v8f acc[4];
#pragma unroll
  for (int g = 0; g < 4; ++g) {
    int n = g * H2D + jcol;
    float bias = bih[n] + bhh[n];
#pragma unroll
    for (int r = 0; r < 8; ++r) acc[g][r] = bias;
  }

#pragma unroll
  for (int kc = 0; kc < 8; ++kc) {
    const _Float16* pa = h1 + mrow * 256 + kc * 32 + hi * 8;
    v16h a = ld_frag(pa, pa + 16);
#pragma unroll
    for (int g = 0; g < 4; ++g) {
      const _Float16* pb = wswz + ((((size_t)(wave * 4 + g) * 8 + kc) * 32 + lane) << 4);
      v16h b = ld_frag(pb, pb + 8);
      acc[g] = wmma16(a, b, acc[g]);
    }
  }

#pragma unroll
  for (int r = 0; r < 8; ++r) {
    int m = hi * 8 + r;
    float ig = sigm(acc[0][r]);
    float gg = tanh_hw(acc[2][r]);
    float og = sigm(acc[3][r]);
    float c  = ig * gg;                 // f*c0 = 0
    h2last[(bm + m) * 128 + 64 + jcol] = og * tanh_hw(c);
  }
}

// ---------------- Final FC: (256,128) @ (4,128)^T + b ----------------
__global__ void k_fc(const float* __restrict__ h2last, const float* __restrict__ wfc,
                     const float* __restrict__ bfc, float* __restrict__ out) {
  int gid = blockIdx.x * blockDim.x + threadIdx.x;
  if (gid >= BATCH * NCLS) return;
  int b = gid >> 2, c = gid & 3;
  float s = bfc[c];
  for (int k = 0; k < 2 * H2D; ++k) s += h2last[b * 128 + k] * wfc[c * 128 + k];
  out[gid] = s;
}

extern "C" void kernel_launch(void* const* d_in, const int* in_sizes, int n_in,
                              void* d_out, int out_size, void* d_ws, size_t ws_size,
                              hipStream_t stream) {
  (void)in_sizes; (void)n_in; (void)out_size; (void)ws_size;
  const float* x     = (const float*)d_in[0];
  const float* wih1f = (const float*)d_in[1];
  const float* whh1f = (const float*)d_in[2];
  const float* bih1f = (const float*)d_in[3];
  const float* bhh1f = (const float*)d_in[4];
  const float* wih1b = (const float*)d_in[5];
  const float* whh1b = (const float*)d_in[6];
  const float* bih1b = (const float*)d_in[7];
  const float* bhh1b = (const float*)d_in[8];
  const float* wih2f = (const float*)d_in[9];
  const float* whh2f = (const float*)d_in[10];
  const float* bih2f = (const float*)d_in[11];
  const float* bhh2f = (const float*)d_in[12];
  const float* wih2b = (const float*)d_in[13];
  const float* bih2b = (const float*)d_in[15];
  const float* bhh2b = (const float*)d_in[16];
  const float* wfc   = (const float*)d_in[17];
  const float* bfc   = (const float*)d_in[18];

  char* ws = (char*)d_ws;
  _Float16* sw_whh1f = (_Float16*)(ws + 0);        // 512x128 f16 = 128 KB
  _Float16* sw_whh1b = (_Float16*)(ws + 131072);
  _Float16* sw_wih2f = (_Float16*)(ws + 262144);   // 256x256 (linear)
  _Float16* sw_wih2b = (_Float16*)(ws + 393216);   // 256x256 (interleaved)
  _Float16* sw_whh2f = (_Float16*)(ws + 524288);   // 256x64  (interleaved)
  _Float16* h1   = (_Float16*)(ws + (1ull << 20));                              // 128 MB
  float*    pre2 = (float*)   (ws + (1ull << 20) + 134217728ull);               // 256 MB
  float*  h2last = (float*)   (ws + (1ull << 20) + 134217728ull + 268435456ull);

  k_swz<<<(512 * 128 + 255) / 256, 256, 0, stream>>>(whh1f, sw_whh1f, 512, 128, 1);
  k_swz<<<(512 * 128 + 255) / 256, 256, 0, stream>>>(whh1b, sw_whh1b, 512, 128, 1);
  k_swz<<<(256 * 256 + 255) / 256, 256, 0, stream>>>(wih2f, sw_wih2f, 256, 256, 0);
  k_swz<<<(256 * 256 + 255) / 256, 256, 0, stream>>>(wih2b, sw_wih2b, 256, 256, 1);
  k_swz<<<(256 * 64 + 255) / 256, 256, 0, stream>>>(whh2f, sw_whh2f, 256, 64, 1);

  k_lstm1<<<dim3(16, 2), 256, 0, stream>>>(x, sw_whh1f, sw_whh1b, wih1f, wih1b,
                                           bih1f, bih1b, bhh1f, bhh1b, h1);
  k_pre2<<<dim3(2048, 8), 256, 0, stream>>>(h1, sw_wih2f, bih2f, bhh2f, pre2);
  k_lstm2_fwd<<<16, 128, 0, stream>>>(pre2, sw_whh2f, h2last);
  k_lstm2_bwd_last<<<16, 128, 0, stream>>>(h1, sw_wih2b, bih2b, bhh2b, h2last);
  k_fc<<<(BATCH * NCLS + 255) / 256, 256, 0, stream>>>(h2last, wfc, bfc, (float*)d_out);
}
